// NN_Decoder_71683004170393
// MI455X (gfx1250) — matrix operand order repaired and checked by reading, not compiled, not run on hardware
//
#include <hip/hip_runtime.h>
#include <hip/hip_bf16.h>

typedef __attribute__((ext_vector_type(16))) _Float16 v16h;
typedef __attribute__((ext_vector_type(8)))  float    v8f;

#define PADL  64
#define EXTRA 256   // padded row width W = T + 256 (64 left, 192 right)

// ---------------- zero ----------------
__global__ void k_zero(float* __restrict__ p, long n) {
  long i = (long)blockIdx.x * blockDim.x + threadIdx.x;
  long stride = (long)gridDim.x * blockDim.x;
  for (; i < n; i += stride) p[i] = 0.f;
}

// ---------------- one-hot routing tables ----------------
__global__ void k_route(const float* __restrict__ Ce, const float* __restrict__ Ci,
                        int* __restrict__ route) {
  int j = blockIdx.x * blockDim.x + threadIdx.x;
  if (j < 500) {
    int r = 0;
    for (int s = 0; s < 20; ++s) if (Ce[s * 500 + j] > 0.5f) r = s;
    route[j] = r;                 // e-synapses -> bins 0..19
  } else if (j < 600) {
    int jj = j - 500;
    int r = 0;
    for (int s = 0; s < 20; ++s) if (Ci[s * 100 + jj] > 0.5f) r = s;
    route[j] = 20 + r;            // i-synapses -> bins 20..39
  }
}

// ---------------- syn_in[40][T] (padded, f32) ----------------
__global__ __launch_bounds__(256) void k_syn(const float* __restrict__ Se,
                                             const float* __restrict__ Si,
                                             const int* __restrict__ route,
                                             float* __restrict__ synp, int T, int W) {
  __shared__ float bins[256 * 40];
  __shared__ int rt[600];
  int tid = threadIdx.x;
  for (int i = tid; i < 600; i += 256) rt[i] = route[i];
  for (int s = 0; s < 40; ++s) bins[tid * 40 + s] = 0.f;
  __syncthreads();
  long t = (long)blockIdx.x * 256 + tid;
  if (t < T) {
    const float* se = Se + t * 500;
    for (int j = 0; j < 500; ++j) {
      float v = se[j];
      if (v != 0.f) bins[tid * 40 + rt[j]] += v;
    }
    const float* si = Si + t * 100;
    for (int j = 0; j < 100; ++j) {
      float v = si[j];
      if (v != 0.f) bins[tid * 40 + rt[500 + j]] += v;
    }
    for (int s = 0; s < 40; ++s)
      synp[(size_t)s * W + PADL + t] = bins[tid * 40 + s];
  }
}

// ---------------- pack kern1 = weight1 @ basis into WMMA A-fragment layout ----
// Apack halves: [mt(2)][s(40)][kc(4)][lane(32)][e(16)]
// A 16x32 f16, contiguous k used here (B matches): lane<16: m=lane, K=e
//                     lane>=16: m=lane-16, K=e+16
__global__ void k_pack_kern1(const float* __restrict__ w1, _Float16* __restrict__ Apack) {
  int idx = blockIdx.x * blockDim.x + threadIdx.x;
  if (idx >= 2 * 40 * 4 * 512) return;
  int e    = idx & 15;
  int lane = (idx >> 4) & 31;
  int kc   = (idx >> 9) & 3;
  int rest = idx >> 11;
  int s    = rest % 40;
  int mt   = rest / 40;
  int m  = lane & 15;
  int kk = (lane < 16) ? e : e + 16;
  int h = mt * 16 + m;
  int k = kc * 32 + kk;
  float val = 0.f;
  if (h < 20 && k < 101) {
    const float* wp = w1 + ((size_t)h * 40 + s) * 34;
    for (int b = 0; b < 34; ++b) {
      float d = (float)(k - 3 * b);
      val += wp[b] * __expf(-(d * d) * (1.0f / 3.0f));
    }
  }
  Apack[idx] = (_Float16)val;
}

// ---------------- kern4 / hist kernel / spike kernel ----------------
__global__ void k_smallprep(const float* __restrict__ w4, const float* __restrict__ Wh,
                            const float* __restrict__ Wspk, const float* __restrict__ Tau,
                            float* __restrict__ kern4, float* __restrict__ hk,
                            float* __restrict__ sk) {
  int i = blockIdx.x * blockDim.x + threadIdx.x;
  if (i < 2020) {
    int h = i / 101, k = i % 101;
    float a = 0.f;
    for (int b = 0; b < 34; ++b) {
      float d = (float)(k - 3 * b);
      a += w4[h * 34 + b] * __expf(-(d * d) * (1.0f / 3.0f));
    }
    kern4[i] = a;
  }
  if (i < 50) {
    float a = 0.f;
    for (int b = 0; b < 18; ++b) {
      float d = (float)(i - 3 * b);
      a += Wh[b] * __expf(-(d * d) * (1.0f / 3.0f));
    }
    hk[i] = a;
    float tau = Tau[0];
    float tt  = (float)i / (tau * tau);
    sk[i] = tt * __expf(-tt) * Wspk[0] * Wspk[0];
  }
}

// ---------------- conv1 via WMMA: out1 = leaky_relu(kern1 * syn windows) ------
// Each wave: 64-column t-strip (4 N-tiles), 2 M-tiles (h 0..15, 16..19+pad),
// K loop over s(40) x kc(4 chunks of 32, k padded 101->128 with zero A).
__global__ __launch_bounds__(128) void k_conv1_wmma(const v16h* __restrict__ Apack,
                                                    const float* __restrict__ synp,
                                                    float* __restrict__ out1p,
                                                    int T, int W) {
  const int lane = threadIdx.x & 31;
  const int wave = blockIdx.x * 4 + (threadIdx.x >> 5);
  const long t0 = (long)wave * 64;
  if (t0 >= T) return;
  v8f acc[4][2];
#pragma unroll
  for (int tl = 0; tl < 4; ++tl)
#pragma unroll
    for (int m = 0; m < 2; ++m)
#pragma unroll
      for (int r = 0; r < 8; ++r) acc[tl][m][r] = 0.f;

  for (int s = 0; s < 40; ++s) {
    const float* srow = synp + (size_t)s * W + PADL + t0 + lane - 50;
    for (int kc = 0; kc < 4; ++kc) {
      v16h A0 = Apack[(size_t)((0 * 40 + s) * 4 + kc) * 32 + lane];
      v16h A1 = Apack[(size_t)((1 * 40 + s) * 4 + kc) * 32 + lane];
      const float* b0 = srow + kc * 32;
#pragma unroll
      for (int tl = 0; tl < 4; ++tl) {
        const float* bp = b0 + tl * 16;
        v16h B;
#pragma unroll
        for (int e = 0; e < 16; ++e) B[e] = (_Float16)bp[e];  // exact: spike counts
        acc[tl][0] = __builtin_amdgcn_wmma_f32_16x16x32_f16(
            false, A0, false, B, (short)0, acc[tl][0], false, false);
        acc[tl][1] = __builtin_amdgcn_wmma_f32_16x16x32_f16(
            false, A1, false, B, (short)0, acc[tl][1], false, false);
      }
    }
  }
  // D layout: VGPR r -> M = r (lanes 0-15) / 8+r (lanes 16-31); N = lane%16
  const int n = lane & 15;
  const int mbase = (lane < 16) ? 0 : 8;
#pragma unroll
  for (int tl = 0; tl < 4; ++tl) {
    long tcol = t0 + tl * 16 + n;
    if (tcol >= T) continue;
#pragma unroll
    for (int r = 0; r < 8; ++r) {
      float v = acc[tl][0][r];
      v = v > 0.f ? v : 0.01f * v;                       // leaky_relu
      out1p[(size_t)(mbase + r) * W + PADL + tcol] = v;  // h < 16 always valid
      int h1 = 16 + mbase + r;
      if (h1 < 20) {
        float w = acc[tl][1][r];
        w = w > 0.f ? w : 0.01f * w;
        out1p[(size_t)h1 * W + PADL + tcol] = w;
      }
    }
  }
}

// ---------------- conv2: out4[t] = sum_{h,k} kern4[h,k] * out1[h, t+k-50] -----
__global__ __launch_bounds__(256) void k_conv2(const float* __restrict__ out1p,
                                               const float* __restrict__ kern4,
                                               float* __restrict__ out4, int T, int W) {
  __shared__ float k4[2020];
  for (int i = threadIdx.x; i < 2020; i += 256) k4[i] = kern4[i];
  __syncthreads();
  long t = (long)blockIdx.x * 256 + threadIdx.x;
  if (t >= T) return;
  float acc = 0.f;
  for (int h = 0; h < 20; ++h) {
    const float* row = out1p + (size_t)h * W + PADL + t - 50;
    const float* kp  = k4 + h * 101;
    for (int k = 0; k < 101; ++k) acc += kp[k] * row[k];
  }
  out4[t] = acc;
}

// ---------------- sequential spike scan (single wave) -------------------------
// contrib ring holds future history contributions: spike at t' adds hk[50-d]
// to slot (t'+d)&63 for d=1..50; p(t) = sigmoid(out4[t] + contrib[t&63] + theta)
__global__ void k_scan(const float* __restrict__ out4, const float* __restrict__ u,
                       const float* __restrict__ hk, const float* __restrict__ theta_p,
                       float* __restrict__ prob, float* __restrict__ spikes, int T) {
  __shared__ float contrib[64];
  const int lane = threadIdx.x;            // 32 threads
  contrib[lane] = 0.f; contrib[lane + 32] = 0.f;
  const float th  = theta_p[0];
  const float hkA = hk[49 - lane];                          // d = lane+1
  const float hkB = (lane <= 17) ? hk[17 - lane] : 0.f;     // d = lane+33
  __syncthreads();
  for (int base = 0; base < T; base += 32) {
    int tl = base + lane;
    float o_v = (tl < T) ? out4[tl] : 0.f;
    float u_v = (tl < T) ? u[tl] : 1.f;
    int jmax = (T - base < 32) ? (T - base) : 32;
    for (int j = 0; j < jmax; ++j) {
      int t = base + j;
      float o_t = __shfl(o_v, j, 32);
      float u_t = __shfl(u_v, j, 32);
      int slot = t & 63;
      float c = contrib[slot];
      if (lane == 0) contrib[slot] = 0.f;     // recycle ring slot
      float x = o_t + c + th;
      float p = 1.f / (1.f + __expf(-x));
      bool spk = u_t < p;                     // bernoulli(p) with given u
      if (lane == 0) { prob[t] = p; spikes[t] = spk ? 1.f : 0.f; }
      if (spk) {
        contrib[(t + 1 + lane) & 63] += hkA;                  // distinct slots
        if (lane <= 17) contrib[(t + 33 + lane) & 63] += hkB; // distinct slots
      }
      __syncthreads();
    }
  }
}

// ---------------- spk_filt[t] = sum_j spk_out[t+j] * sk[49-j] -----------------
__global__ __launch_bounds__(256) void k_spkfilt(const float* __restrict__ spikes,
                                                 const float* __restrict__ sk,
                                                 float* __restrict__ outf, int T) {
  __shared__ float kk[50];
  if (threadIdx.x < 50) kk[threadIdx.x] = sk[threadIdx.x];
  __syncthreads();
  long t = (long)blockIdx.x * 256 + threadIdx.x;
  if (t >= T) return;
  float acc = 0.f;
  for (int j = 0; j < 50; ++j) {
    long idx = t + j - 50;                       // spk_out = [zeros(50), spikes]
    float v = (idx >= 0) ? spikes[idx] : 0.f;
    acc += v * kk[49 - j];
  }
  outf[t] = acc;
}

extern "C" void kernel_launch(void* const* d_in, const int* in_sizes, int n_in,
                              void* d_out, int out_size, void* d_ws, size_t ws_size,
                              hipStream_t stream) {
  (void)n_in; (void)out_size; (void)ws_size;
  const float* Se  = (const float*)d_in[0];
  const float* Si  = (const float*)d_in[1];
  const float* Ce  = (const float*)d_in[2];
  const float* Ci  = (const float*)d_in[3];
  const float* w1  = (const float*)d_in[4];
  const float* w4  = (const float*)d_in[5];
  const float* Wh  = (const float*)d_in[6];
  const float* Th  = (const float*)d_in[7];
  const float* Ws  = (const float*)d_in[8];
  const float* Tau = (const float*)d_in[9];
  const float* u   = (const float*)d_in[10];
  const int T = in_sizes[10];       // 200000
  const int W = T + EXTRA;

  char* p = (char*)d_ws;
  auto carve = [&p](size_t bytes) {
    void* r = (void*)p;
    p += (bytes + 255) & ~((size_t)255);
    return r;
  };
  int*      route  = (int*)      carve(600 * sizeof(int));
  float*    synp   = (float*)    carve((size_t)40 * W * sizeof(float));
  float*    out1p  = (float*)    carve((size_t)20 * W * sizeof(float));
  _Float16* Apack  = (_Float16*) carve((size_t)2 * 40 * 4 * 512 * 2);
  float*    kern4  = (float*)    carve(2020 * sizeof(float));
  float*    hk     = (float*)    carve(64 * sizeof(float));
  float*    sk     = (float*)    carve(64 * sizeof(float));
  float*    out4   = (float*)    carve((size_t)T * sizeof(float));
  float*    spikes = (float*)    carve((size_t)T * sizeof(float));
  float*    spkf   = (float*)d_out;        // output 0: spk_filt [T]
  float*    prob   = (float*)d_out + T;    // output 1: prob_out [T]

  k_zero<<<2048, 256, 0, stream>>>(synp, (long)40 * W);
  k_zero<<<2048, 256, 0, stream>>>(out1p, (long)20 * W);
  k_route<<<3, 256, 0, stream>>>(Ce, Ci, route);
  k_syn<<<(T + 255) / 256, 256, 0, stream>>>(Se, Si, route, synp, T, W);
  k_pack_kern1<<<640, 256, 0, stream>>>(w1, Apack);
  k_smallprep<<<8, 256, 0, stream>>>(w4, Wh, Ws, Tau, kern4, hk, sk);
  int waves = (T + 63) / 64;
  k_conv1_wmma<<<(waves + 3) / 4, 128, 0, stream>>>((const v16h*)Apack, synp, out1p, T, W);
  k_conv2<<<(T + 255) / 256, 256, 0, stream>>>(out1p, kern4, out4, T, W);
  k_scan<<<1, 32, 0, stream>>>(out4, u, hk, Th, prob, spikes, T);
  k_spkfilt<<<(T + 255) / 256, 256, 0, stream>>>(spikes, sk, spkf, T);
}
